// MultiHeadSelfAttention_19413252178032
// MI455X (gfx1250) — compile-verified, matrix operand order checked by reading, not run on hardware
//
#include <hip/hip_runtime.h>

typedef __attribute__((ext_vector_type(16))) _Float16 v16h;
typedef __attribute__((ext_vector_type(8)))  float    v8f;

#define D_MODEL 1024
#define N_HEADS 16
#define D_HEAD  64
#define BATCH   2
#define SEQ     2048
#define BT      (BATCH * SEQ)     // 4096 token rows
#define QKV_N   (3 * D_MODEL)     // 3072

#define GEMM_BM  32               // rows per block (shared A tile)
#define GEMM_BN  512              // 8 waves x 64 cols
#define ASTRIDE  40               // LDS row stride in halves (80B: conflict-free b128 reads)

// ---------------------------------------------------------------------------
// gfx1250 async copy: global -> LDS, tracked by ASYNCcnt (no VGPR round trip)
// ---------------------------------------------------------------------------
__device__ inline void async_load_b128(void* lds, const void* gptr) {
  const unsigned l = (unsigned)(unsigned long long)lds;  // low 32 bits = LDS offset
  asm volatile("global_load_async_to_lds_b128 %0, %1, off"
               :: "v"(l), "v"(gptr) : "memory");
}

#if __has_builtin(__builtin_amdgcn_s_wait_asynccnt)
#define WAIT_ASYNC(n) __builtin_amdgcn_s_wait_asynccnt(n)
#else
#define WAIT_ASYNC(n) asm volatile("s_wait_asynccnt %0" :: "i"(n) : "memory")
#endif

// ---------------------------------------------------------------------------
// WMMA fragment helpers (wave32, 16x16x32 f16 -> f32)
//  A-frag (16x32 MxK): lane r=lane&15 holds row M=r; off = (lane>=16)?8:0
//      a[i] = A(r, off+i), a[8+i] = A(r, off+16+i)
//  B-frag (32x16 KxN): lane c=lane&15 holds col N=c, same K split
//  C/D (16x16 f32): lane holds col N=lane&15; c[i] = row M = i + 8*(lane>=16)
// All sources arranged so every fragment load is two contiguous 16B runs.
// ---------------------------------------------------------------------------

__device__ inline v8f wmma_f16(v16h a, v16h b, v8f c) {
  return __builtin_amdgcn_wmma_f32_16x16x32_f16(
      /*neg_a=*/false, a, /*neg_b=*/false, b,
      /*c_mod=*/(short)0, c, /*reuse_a=*/false, /*reuse_b=*/false);
}

// A-frag from row-major global src
__device__ inline v16h load_fragA(const _Float16* __restrict__ src, int ld,
                                  int m0, int k0, int lane) {
  const int r   = lane & 15;
  const int off = (lane & 16) ? 8 : 0;
  const _Float16* p = src + (size_t)(m0 + r) * ld + k0 + off;
  v16h a;
#pragma unroll
  for (int i = 0; i < 8; ++i) { a[i] = p[i]; a[8 + i] = p[16 + i]; }
  return a;
}

// A-frag from the padded LDS tile (rows m0..m0+15 of the 32-row tile)
__device__ inline v16h load_fragA_lds(const _Float16* buf, int m0, int lane) {
  const int r   = lane & 15;
  const int off = (lane & 16) ? 8 : 0;
  const _Float16* p = buf + (m0 + r) * ASTRIDE + off;
  v16h a;
#pragma unroll
  for (int i = 0; i < 8; ++i) { a[i] = p[i]; a[8 + i] = p[16 + i]; }
  return a;
}

// B-frag from column-major src (src holds B^T row-major)
__device__ inline v16h load_fragB_cm(const _Float16* __restrict__ src, int ld,
                                     int k0, int n0, int lane) {
  const int c   = lane & 15;
  const int off = (lane & 16) ? 8 : 0;
  const _Float16* p = src + (size_t)(n0 + c) * ld + k0 + off;
  v16h b;
#pragma unroll
  for (int i = 0; i < 8; ++i) { b[i] = p[i]; b[8 + i] = p[16 + i]; }
  return b;
}

// ---------------------------------------------------------------------------
// Stage 0a: straight fp32 -> f16
// ---------------------------------------------------------------------------
__global__ void f32_to_f16_kernel(const float* __restrict__ in,
                                  _Float16* __restrict__ out, int n) {
  int i = blockIdx.x * blockDim.x + threadIdx.x;
  int stride = gridDim.x * blockDim.x;
  for (; i < n; i += stride) out[i] = (_Float16)in[i];
}

// Stage 0b: fp32 [1024 x C] row-major -> f16 transposed [C x 1024]
__global__ void f32_to_f16_T_kernel(const float* __restrict__ in,
                                    _Float16* __restrict__ out, int n, int C) {
  int i = blockIdx.x * blockDim.x + threadIdx.x;
  int stride = gridDim.x * blockDim.x;
  for (; i < n; i += stride) {
    const int r = i & (D_MODEL - 1);
    const int c = i >> 10;
    out[i] = (_Float16)in[(size_t)r * C + c];
  }
}

// ---------------------------------------------------------------------------
// Shared-A GEMM core: C[32 x 512 strip] = A[32 x 1024] @ B^T-stored weights.
// A tile double-buffered in LDS via async copies; B frags direct b128 loads.
// ---------------------------------------------------------------------------
struct GemmAcc { v8f a[2][4]; };

__device__ inline GemmAcc gemm_core(const _Float16* __restrict__ Asrc,
                                    const _Float16* __restrict__ wT,
                                    _Float16* abuf,   // [2][GEMM_BM*ASTRIDE]
                                    int m0, int n0, int tid, int lane) {
  const int srow = tid >> 2;          // tid<128: rows 0..31
  const int scol = (tid & 3) * 8;     // 4 x 16B segments per 64B row

  if (tid < 128)
    async_load_b128(abuf + srow * ASTRIDE + scol,
                    Asrc + (size_t)(m0 + srow) * D_MODEL + scol);

  GemmAcc r;
#pragma unroll
  for (int s = 0; s < 2; ++s)
#pragma unroll
    for (int j = 0; j < 4; ++j) r.a[s][j] = (v8f){};

  int buf = 0;
  for (int k0 = 0; k0 < D_MODEL; k0 += 32, buf ^= 1) {
    const bool has_next = (k0 + 32 < D_MODEL);
    if (has_next && tid < 128)
      async_load_b128(abuf + (buf ^ 1) * (GEMM_BM * ASTRIDE) + srow * ASTRIDE + scol,
                      Asrc + (size_t)(m0 + srow) * D_MODEL + k0 + 32 + scol);
    if (has_next) WAIT_ASYNC(1); else WAIT_ASYNC(0);
    __syncthreads();                       // staged tile visible block-wide

    const _Float16* cur = abuf + buf * (GEMM_BM * ASTRIDE);
    v16h a0 = load_fragA_lds(cur, 0,  lane);
    v16h a1 = load_fragA_lds(cur, 16, lane);
#pragma unroll
    for (int j = 0; j < 4; ++j) {
      v16h b = load_fragB_cm(wT, D_MODEL, k0, n0 + j * 16, lane);
      r.a[0][j] = wmma_f16(a0, b, r.a[0][j]);
      r.a[1][j] = wmma_f16(a1, b, r.a[1][j]);
    }
    __syncthreads();                       // done reading before next overwrite
  }
  return r;
}

// ---------------------------------------------------------------------------
// Stage 1: QKV GEMM -> Q,K as [B,H,T,64] (Q pre-scaled by 1/8), V as [B,H,64,T]
// ---------------------------------------------------------------------------
__global__ __launch_bounds__(256) void qkv_gemm_kernel(
    const _Float16* __restrict__ xh,
    const _Float16* __restrict__ wT,      // [3072][1024]
    _Float16* __restrict__ Qa,
    _Float16* __restrict__ Ka,
    _Float16* __restrict__ Vt) {
  __shared__ _Float16 abuf[2][GEMM_BM * ASTRIDE];
  const int tid  = threadIdx.x;
  const int lane = tid & 31;
  const int wave = tid >> 5;
  const int NB = QKV_N / GEMM_BN;        // 6 block strips
  const int m0 = (blockIdx.x / NB) * GEMM_BM;
  const int n0 = (blockIdx.x % NB) * GEMM_BN + wave * 64;

  GemmAcc acc = gemm_core(xh, wT, &abuf[0][0], m0, n0, tid, lane);

  // Epilogue: region and head are wave-uniform (n0 is a multiple of 64).
  const int region = n0 >> 10;           // 0=q 1=k 2=v
  const int h      = (n0 & (D_MODEL - 1)) >> 6;
  const int hi = (lane & 16) ? 8 : 0;
  const int cn = lane & 15;

  if (region == 0) {
#pragma unroll
    for (int s = 0; s < 2; ++s)
#pragma unroll
      for (int j = 0; j < 4; ++j)
#pragma unroll
        for (int i = 0; i < 8; ++i) {
          const int row = m0 + s * 16 + i + hi;
          const int bb = row >> 11, t = row & (SEQ - 1);
          const int d = j * 16 + cn;
          Qa[((((size_t)bb * N_HEADS + h) * SEQ) + t) * D_HEAD + d] =
              (_Float16)(acc.a[s][j][i] * 0.125f);
        }
  } else if (region == 1) {
#pragma unroll
    for (int s = 0; s < 2; ++s)
#pragma unroll
      for (int j = 0; j < 4; ++j)
#pragma unroll
        for (int i = 0; i < 8; ++i) {
          const int row = m0 + s * 16 + i + hi;
          const int bb = row >> 11, t = row & (SEQ - 1);
          const int d = j * 16 + cn;
          Ka[((((size_t)bb * N_HEADS + h) * SEQ) + t) * D_HEAD + d] =
              (_Float16)acc.a[s][j][i];
        }
  } else {
#pragma unroll
    for (int s = 0; s < 2; ++s)
#pragma unroll
      for (int j = 0; j < 4; ++j)
#pragma unroll
        for (int i = 0; i < 8; ++i) {
          const int row = m0 + s * 16 + i + hi;
          const int bb = row >> 11, t = row & (SEQ - 1);
          const int d = j * 16 + cn;
          Vt[((((size_t)bb * N_HEADS + h) * D_HEAD) + d) * SEQ + t] =
              (_Float16)acc.a[s][j][i];       // coalesced in t
        }
  }
}

// ---------------------------------------------------------------------------
// Stage 2: causal flash attention, transposed formulation (unchanged core).
// ---------------------------------------------------------------------------
__global__ void attn_kernel(const _Float16* __restrict__ Qa,
                            const _Float16* __restrict__ Ka,
                            const _Float16* __restrict__ Vt,
                            _Float16* __restrict__ ctx) {
  const int lane = threadIdx.x & 31;
  const int wid  = (blockIdx.x * blockDim.x + threadIdx.x) >> 5;
  const int qt = wid & 127;
  const int h  = (wid >> 7) & (N_HEADS - 1);
  const int b  = wid >> 11;
  const int qbase = qt * 16;

  const _Float16* Qh = Qa + (((size_t)b * N_HEADS + h) * SEQ) * D_HEAD;
  const _Float16* Kh = Ka + (((size_t)b * N_HEADS + h) * SEQ) * D_HEAD;
  const _Float16* Vh = Vt + (((size_t)b * N_HEADS + h) * D_HEAD) * SEQ;

  const v16h qb0 = load_fragB_cm(Qh, D_HEAD, 0,  qbase, lane);
  const v16h qb1 = load_fragB_cm(Qh, D_HEAD, 32, qbase, lane);

  v8f o[4] = {};
  float m = -1e30f, l = 0.0f;

  const int q    = qbase + (lane & 15);
  const int off8 = (lane & 16) ? 8 : 0;
  const int kend = qbase + 16;

  for (int kb = 0; kb < kend; kb += 32) {
    if (kb + 32 < kend) {
      __builtin_prefetch(Kh + (size_t)(kb + 32 + (lane & 15)) * D_HEAD, 0, 3);
      __builtin_prefetch(Vh + (size_t)(lane & 15) * SEQ + kb + 32, 0, 3);
    }
    v8f s0 = {}, s1 = {};
    {
      v16h ka = load_fragA(Kh, D_HEAD, kb, 0, lane);
      v16h kc = load_fragA(Kh, D_HEAD, kb, 32, lane);
      s0 = wmma_f16(ka, qb0, s0);
      s0 = wmma_f16(kc, qb1, s0);
    }
    {
      v16h ka = load_fragA(Kh, D_HEAD, kb + 16, 0, lane);
      v16h kc = load_fragA(Kh, D_HEAD, kb + 16, 32, lane);
      s1 = wmma_f16(ka, qb0, s1);
      s1 = wmma_f16(kc, qb1, s1);
    }

    if (kb + 31 > qbase) {
      const int k0 = kb + off8;
#pragma unroll
      for (int i = 0; i < 8; ++i) {
        if (k0 + i      > q) s0[i] = -1e30f;
        if (k0 + 16 + i > q) s1[i] = -1e30f;
      }
    }

    float mx = s0[0];
#pragma unroll
    for (int i = 0; i < 8; ++i) { mx = fmaxf(mx, s0[i]); mx = fmaxf(mx, s1[i]); }
    mx = fmaxf(mx, __shfl_xor(mx, 16, 32));
    const float mn    = fmaxf(m, mx);
    const float alpha = __expf(m - mn);
    m = mn;

    float rsum = 0.0f;
    v16h p;
#pragma unroll
    for (int i = 0; i < 8; ++i) {
      const float p0 = __expf(s0[i] - mn);
      const float p1 = __expf(s1[i] - mn);
      rsum += p0 + p1;
      p[i]     = (_Float16)p0;
      p[8 + i] = (_Float16)p1;
    }
    rsum += __shfl_xor(rsum, 16, 32);
    l = l * alpha + rsum;

#pragma unroll
    for (int d = 0; d < 4; ++d) {
#pragma unroll
      for (int i = 0; i < 8; ++i) o[d][i] *= alpha;
      v16h va = load_fragA(Vh, SEQ, d * 16, kb, lane);
      o[d] = wmma_f16(va, p, o[d]);
    }
  }

  const float inv = 1.0f / l;
  _Float16* crow = ctx + ((size_t)(b * SEQ + q)) * D_MODEL + h * D_HEAD;
#pragma unroll
  for (int d = 0; d < 4; ++d)
#pragma unroll
    for (int i = 0; i < 8; ++i)
      crow[d * 16 + off8 + i] = (_Float16)(o[d][i] * inv);
}

// ---------------------------------------------------------------------------
// Stage 3: output projection -> fp32 d_out
// ---------------------------------------------------------------------------
__global__ __launch_bounds__(256) void proj_gemm_kernel(
    const _Float16* __restrict__ ctxh,
    const _Float16* __restrict__ wT,      // [1024][1024]
    float* __restrict__ out) {
  __shared__ _Float16 abuf[2][GEMM_BM * ASTRIDE];
  const int tid  = threadIdx.x;
  const int lane = tid & 31;
  const int wave = tid >> 5;
  const int NB = D_MODEL / GEMM_BN;      // 2 block strips
  const int m0 = (blockIdx.x / NB) * GEMM_BM;
  const int n0 = (blockIdx.x % NB) * GEMM_BN + wave * 64;

  GemmAcc acc = gemm_core(ctxh, wT, &abuf[0][0], m0, n0, tid, lane);

  const int hi = (lane & 16) ? 8 : 0;
  const int cn = lane & 15;
#pragma unroll
  for (int s = 0; s < 2; ++s)
#pragma unroll
    for (int j = 0; j < 4; ++j)
#pragma unroll
      for (int i = 0; i < 8; ++i)
        out[(size_t)(m0 + s * 16 + i + hi) * D_MODEL + n0 + j * 16 + cn] =
            acc.a[s][j][i];
}

// ---------------------------------------------------------------------------
// Launch
// ---------------------------------------------------------------------------
extern "C" void kernel_launch(void* const* d_in, const int* in_sizes, int n_in,
                              void* d_out, int out_size, void* d_ws, size_t ws_size,
                              hipStream_t stream) {
  const float* x      = (const float*)d_in[0];
  const float* w_qkv  = (const float*)d_in[1];
  const float* w_proj = (const float*)d_in[2];
  float* out = (float*)d_out;

  _Float16* xh      = (_Float16*)d_ws;
  _Float16* wqkvT   = xh      + (size_t)BT * D_MODEL;        // [3072][1024]
  _Float16* wprojT  = wqkvT   + (size_t)D_MODEL * QKV_N;     // [1024][1024]
  _Float16* Qa      = wprojT  + (size_t)D_MODEL * D_MODEL;   // [B,H,T,64]
  _Float16* Ka      = Qa      + (size_t)BT * D_MODEL;        // [B,H,T,64]
  _Float16* Vt      = Ka      + (size_t)BT * D_MODEL;        // [B,H,64,T]
  _Float16* ctxh    = Vt      + (size_t)BT * D_MODEL;        // [BT, D]

  const int n_x  = BT * D_MODEL;
  const int n_wq = D_MODEL * QKV_N;
  const int n_wp = D_MODEL * D_MODEL;
  f32_to_f16_kernel<<<(n_x + 255) / 256, 256, 0, stream>>>(x, xh, n_x);
  f32_to_f16_T_kernel<<<(n_wq + 255) / 256, 256, 0, stream>>>(w_qkv, wqkvT, n_wq, QKV_N);
  f32_to_f16_T_kernel<<<(n_wp + 255) / 256, 256, 0, stream>>>(w_proj, wprojT, n_wp, D_MODEL);

  // QKV: (4096/32) * (3072/512) = 768 blocks of 8 waves
  qkv_gemm_kernel<<<768, 256, 0, stream>>>(xh, wqkvT, Qa, Ka, Vt);

  // Attention: B*H*(T/16) = 4096 waves
  attn_kernel<<<4096 / 8, 256, 0, stream>>>(Qa, Ka, Vt, ctxh);

  // Projection: (4096/32) * (1024/512) = 256 blocks
  proj_gemm_kernel<<<256, 256, 0, stream>>>(ctxh, wprojT, out);
}
